// SelectiveSSM_27848567947560
// MI455X (gfx1250) — matrix operand order repaired and checked by reading, not compiled, not run on hardware
//
#include <hip/hip_runtime.h>
#include <hip/hip_bf16.h>
#include <math.h>

// ---- problem constants (match reference) ----
#define D_MODEL 1024
#define D_STATE 16
#define D_CONV  4
#define D_INNER 2048          // D_MODEL * EXPAND
#define BATCH   2
#define SEQ     2048
#define NTOK    (BATCH * SEQ) // 4096 tokens
#define K_SSM   (2 * D_STATE + 1) // 33

typedef __attribute__((ext_vector_type(16))) __bf16 v16bf;
typedef __attribute__((ext_vector_type(8)))  __bf16 bf16x8;
typedef __attribute__((ext_vector_type(4)))  __bf16 bf16x4;
typedef __attribute__((ext_vector_type(8)))  float  v8f;

#define LDSPAD 40   // row stride (elements): 80B -> 16B-aligned cols, conflict-free

// ---------------------------------------------------------------------------
// WMMA GEMM:  C[M,N] = A[M,K] * W[N,K]^T   (fp32 in memory, bf16 WMMA, f32 acc)
//
// Block = 256 threads = 8 waves arranged 2 (M) x 4 (N).
// Block tile = 128 (M) x 128 (N); wave tile = 64 x 32 -> acc[4][2] v8f,
// 8 x v_wmma_f32_16x16x32_bf16 per 32-wide K-step.
// A and B tiles (128x32 each) are staged in LDS as bf16 per K-step with
// float4 global loads + packed bf16x4 LDS stores.
//
// Fragment layouts used here (wave32, 16-bit operands), lane = h*16 + r:
//   A (16x32 MxK): row = r; element j holds K = h*16 + j
//                  -> one 16-elem run at LDS cols [h*16, +16)
//   B (32x16 KxN): col = r; element e holds K = h*16 + e
//                  -> one 16-elem run at LDS cols [h*16, +16)
//   C (16x16 f32): acc[v] is (M = v + 8*h, N = r)
// ---------------------------------------------------------------------------
__global__ void __launch_bounds__(256)
wmma_gemm_nt(const float* __restrict__ A, const float* __restrict__ W,
             float* __restrict__ C, int M, int N, int K)
{
    __shared__ __bf16 As[128][LDSPAD];
    __shared__ __bf16 Bs[128][LDSPAD];

    const int tid  = threadIdx.x;
    const int wv   = tid >> 5;
    const int lane = tid & 31;
    const int h    = lane >> 4;
    const int r    = lane & 15;
    const int wm   = (wv >> 2) * 64;   // wave M offset in block tile
    const int wn   = (wv & 3) * 32;    // wave N offset in block tile

    const int m0 = blockIdx.y * 128;
    const int n0 = blockIdx.x * 128;

    // static staging map: 2 threads per row, 16 floats (4x float4) each
    const int srow = tid >> 1;
    const int scol = (tid & 1) << 4;
    const float* Abase = A + (size_t)(m0 + srow) * K + scol;
    const float* Bbase = W + (size_t)(n0 + srow) * K + scol;

    v8f acc[4][2] = {};

    for (int k0 = 0; k0 < K; k0 += 32) {
        __syncthreads();   // previous tile fully consumed

        // ---- stage A and B tiles (fp32 -> bf16) ----
        {
            const float4* ap = (const float4*)(Abase + k0);
            const float4* bp = (const float4*)(Bbase + k0);
#pragma unroll
            for (int q = 0; q < 4; ++q) {
                float4 fa = ap[q];
                float4 fb = bp[q];
                bf16x4 pa = { (__bf16)fa.x, (__bf16)fa.y, (__bf16)fa.z, (__bf16)fa.w };
                bf16x4 pb = { (__bf16)fb.x, (__bf16)fb.y, (__bf16)fb.z, (__bf16)fb.w };
                *(bf16x4*)&As[srow][scol + 4 * q] = pa;
                *(bf16x4*)&Bs[srow][scol + 4 * q] = pb;
            }
            if (k0 + 32 < K) {
                __builtin_prefetch(Abase + k0 + 32, 0, 0);  // global_prefetch_b8
                __builtin_prefetch(Bbase + k0 + 32, 0, 0);
            }
        }
        __syncthreads();

        // ---- B fragments (2 per wave) ----
        v16bf bfr[2];
#pragma unroll
        for (int ni = 0; ni < 2; ++ni) {
            bf16x8 lo = *(const bf16x8*)&Bs[wn + ni * 16 + r][h * 16];
            bf16x8 hi = *(const bf16x8*)&Bs[wn + ni * 16 + r][h * 16 + 8];
            bfr[ni] = __builtin_shufflevector(lo, hi, 0, 1, 2, 3, 4, 5, 6, 7,
                                              8, 9, 10, 11, 12, 13, 14, 15);
        }
        // ---- A fragments (4 per wave) ----
        v16bf afr[4];
#pragma unroll
        for (int mi = 0; mi < 4; ++mi) {
            bf16x8 lo = *(const bf16x8*)&As[wm + mi * 16 + r][h * 16];
            bf16x8 hi = *(const bf16x8*)&As[wm + mi * 16 + r][h * 16 + 8];
            afr[mi] = __builtin_shufflevector(lo, hi, 0, 1, 2, 3, 4, 5, 6, 7,
                                              8, 9, 10, 11, 12, 13, 14, 15);
        }

        // ---- 8 WMMAs ----
#pragma unroll
        for (int mi = 0; mi < 4; ++mi)
#pragma unroll
            for (int ni = 0; ni < 2; ++ni)
                acc[mi][ni] = __builtin_amdgcn_wmma_f32_16x16x32_bf16(
                    false, afr[mi], false, bfr[ni],
                    (short)0, acc[mi][ni], false, false);
    }

    // ---- store 64x32 wave tile ----
#pragma unroll
    for (int mi = 0; mi < 4; ++mi)
#pragma unroll
        for (int ni = 0; ni < 2; ++ni)
#pragma unroll
            for (int v = 0; v < 8; ++v)
                C[(size_t)(m0 + wm + mi * 16 + v + 8 * h) * N
                  + n0 + wn + ni * 16 + r] = acc[mi][ni][v];
}

// ---------------------------------------------------------------------------
// Depthwise causal conv (width 4) + bias + SiLU on the x half of xz.
// ---------------------------------------------------------------------------
__global__ void __launch_bounds__(256)
conv_silu_kernel(const float* __restrict__ xz, const float* __restrict__ conv_w,
                 const float* __restrict__ conv_b, float* __restrict__ x_d)
{
    const size_t idx = (size_t)blockIdx.x * blockDim.x + threadIdx.x;
    if (idx >= (size_t)NTOK * D_INNER) return;
    const int d  = (int)(idx % D_INNER);
    const int bt = (int)(idx / D_INNER);
    const int t  = bt % SEQ;

    float acc = conv_b[d];
#pragma unroll
    for (int j = 0; j < D_CONV; ++j) {
        const int tt = t - (D_CONV - 1) + j;
        if (tt >= 0)
            acc += xz[(size_t)(bt + (tt - t)) * (2 * D_INNER) + d] * conv_w[d * D_CONV + j];
    }
    x_d[idx] = acc / (1.0f + __expf(-acc));   // SiLU
}

// ---------------------------------------------------------------------------
// ssm_in[token, 0..32] = x_d[token,:] . W_x[k,:]   (one wave per (token,k))
// ---------------------------------------------------------------------------
__global__ void __launch_bounds__(256)
ssm_proj_kernel(const float* __restrict__ x_d, const float* __restrict__ W_x,
                float* __restrict__ ssm_in)
{
    const int gw   = (int)(((size_t)blockIdx.x * blockDim.x + threadIdx.x) >> 5);
    const int lane = threadIdx.x & 31;
    if (gw >= NTOK * K_SSM) return;
    const int tok = gw / K_SSM;
    const int ko  = gw % K_SSM;

    const float* xp = x_d + (size_t)tok * D_INNER;
    const float* wp = W_x + (size_t)ko  * D_INNER;
    float s = 0.f;
    for (int k = lane; k < D_INNER; k += 32) s += xp[k] * wp[k];
#pragma unroll
    for (int o = 16; o; o >>= 1) s += __shfl_down(s, o, 32);
    if (lane == 0) ssm_in[(size_t)tok * K_SSM + ko] = s;
}

// ---------------------------------------------------------------------------
// Sequential selective scan. One thread per (channel d, state s); a channel's
// 16 state lanes are contiguous -> inside one wave, so the in-place y-over-x_d
// write (lane s==0) cannot race the same-iteration u/z loads.
// ---------------------------------------------------------------------------
__global__ void __launch_bounds__(256)
scan_kernel(const float* __restrict__ xz, float* __restrict__ x_d_io,
            const float* __restrict__ ssm_in, const float* __restrict__ W_dt,
            const float* __restrict__ b_dt, const float* __restrict__ A_log,
            const float* __restrict__ D_param)
{
    const int s    = threadIdx.x & 15;
    const int dloc = threadIdx.x >> 4;
    const int b    = blockIdx.x >> 7;
    const int d    = ((blockIdx.x & 127) << 4) + dloc;

    const float Acoef = -__expf(A_log[d * D_STATE + s]);
    const float wdt   = W_dt[d];
    const float bdt   = b_dt[d];
    const float Dp    = D_param[d];

    float h = 0.f;

    for (int t = 0; t < SEQ; ++t) {
        const size_t bt = (size_t)b * SEQ + t;
        const float* si = ssm_in + bt * K_SSM;

        const float dtr = si[0];
        const float Bv  = si[1 + s];
        const float Cv  = si[1 + D_STATE + s];
        const float u   = x_d_io[bt * D_INNER + d];
        const float z   = xz[bt * (2 * D_INNER) + D_INNER + d];

        const float pre   = dtr * wdt + bdt;
        const float delta = (pre > 20.f) ? pre : log1pf(__expf(pre));

        h = __expf(delta * Acoef) * h + delta * Bv * u;

        float val = h * Cv;
#pragma unroll
        for (int o = 8; o; o >>= 1) val += __shfl_xor(val, o, 16);

        if (s == 0) {
            const float y  = val + Dp * u;
            const float sz = z / (1.0f + __expf(-z));
            x_d_io[bt * D_INNER + d] = y * sz;
        }
    }
}

// ---------------------------------------------------------------------------
extern "C" void kernel_launch(void* const* d_in, const int* in_sizes, int n_in,
                              void* d_out, int out_size, void* d_ws, size_t ws_size,
                              hipStream_t stream)
{
    const float* x       = (const float*)d_in[0];
    const float* W_in    = (const float*)d_in[1];
    const float* conv_w  = (const float*)d_in[2];
    const float* conv_b  = (const float*)d_in[3];
    const float* W_x     = (const float*)d_in[4];
    const float* W_dt    = (const float*)d_in[5];
    const float* b_dt    = (const float*)d_in[6];
    const float* A_log   = (const float*)d_in[7];
    const float* D_param = (const float*)d_in[8];
    const float* W_out   = (const float*)d_in[9];
    float*       out     = (float*)d_out;

    // workspace layout (fp32): xz | x_d (later y) | ssm_in   ~101 MB
    float* xz  = (float*)d_ws;
    float* x_d = xz  + (size_t)NTOK * 2 * D_INNER;
    float* ssm = x_d + (size_t)NTOK * D_INNER;

    // 1) xz = x . W_in^T        M=4096  N=4096  K=1024
    wmma_gemm_nt<<<dim3((2 * D_INNER) / 128, NTOK / 128), 256, 0, stream>>>(
        x, W_in, xz, NTOK, 2 * D_INNER, D_MODEL);

    // 2) depthwise conv + SiLU -> x_d
    conv_silu_kernel<<<(NTOK * D_INNER) / 256, 256, 0, stream>>>(
        xz, conv_w, conv_b, x_d);

    // 3) ssm_in = x_d . W_x^T
    {
        const size_t waves = (size_t)NTOK * K_SSM;
        const int blocks = (int)((waves * 32 + 255) / 256);
        ssm_proj_kernel<<<blocks, 256, 0, stream>>>(x_d, W_x, ssm);
    }

    // 4) selective scan (y in place over x_d)
    scan_kernel<<<(BATCH * D_INNER) / 16, 256, 0, stream>>>(
        xz, x_d, ssm, W_dt, b_dt, A_log, D_param);

    // 5) out = y . W_out^T      M=4096  N=1024  K=2048
    wmma_gemm_nt<<<dim3(D_MODEL / 128, NTOK / 128), 256, 0, stream>>>(
        x_d, W_out, out, NTOK, D_MODEL, D_INNER);
}
